// PointNet2FeatureExtractorWithFP_52304111730782
// MI455X (gfx1250) — matrix operand order repaired and checked by reading, not compiled, not run on hardware
//
#include <hip/hip_runtime.h>
#include <hip/hip_bf16.h>

typedef __attribute__((ext_vector_type(16))) __bf16 v16bf;
typedef __attribute__((ext_vector_type(8)))  __bf16 v8bf;
typedef __attribute__((ext_vector_type(8)))  float  v8f;

#define EW_BLOCKS 2048
#define FINF 3.4e38f

__device__ __forceinline__ unsigned short f2bf(float f) {
    unsigned int u = __float_as_uint(f);
    unsigned int r = u + 0x7FFFu + ((u >> 16) & 1u);   // round-to-nearest-even
    return (unsigned short)(r >> 16);
}

// ---------------------------------------------------------------------------
// Farthest point sampling: one block per batch, mind[] resident in LDS.
// Emits new_xyz[it] = xyz[last] with last_0 = 0 (matches reference scan).
// ---------------------------------------------------------------------------
__global__ __launch_bounds__(1024) void fps_kernel(const float* __restrict__ xyz,
                                                   int n, int S,
                                                   float* __restrict__ nxyz) {
    __shared__ float mind[8192];
    __shared__ float rv[1024];
    __shared__ int   ri[1024];
    const int b   = blockIdx.x;
    const int tid = threadIdx.x;
    const float* px = xyz + (size_t)b * n * 3;
    for (int i = tid; i < n; i += 1024) mind[i] = FINF;
    __syncthreads();
    int last = 0;
    for (int it = 0; it < S; ++it) {
        float cx = px[last * 3 + 0], cy = px[last * 3 + 1], cz = px[last * 3 + 2];
        if (tid == 0) {
            float* o = nxyz + ((size_t)b * S + it) * 3;
            o[0] = cx; o[1] = cy; o[2] = cz;
        }
        float bv = -1.f; int bi = 0;
        for (int i = tid; i < n; i += 1024) {
            float dx = px[i * 3 + 0] - cx;
            float dy = px[i * 3 + 1] - cy;
            float dz = px[i * 3 + 2] - cz;
            float d  = dx * dx + dy * dy + dz * dz;
            float m  = fminf(mind[i], d);
            mind[i]  = m;
            if (m > bv) { bv = m; bi = i; }
        }
        rv[tid] = bv; ri[tid] = bi;
        __syncthreads();
        for (int s2 = 512; s2 > 0; s2 >>= 1) {
            if (tid < s2) {
                float ov = rv[tid + s2]; int oi = ri[tid + s2];
                if (ov > rv[tid] || (ov == rv[tid] && oi < ri[tid])) { rv[tid] = ov; ri[tid] = oi; }
            }
            __syncthreads();
        }
        last = ri[0];
        __syncthreads();
    }
}

// ---------------------------------------------------------------------------
// Ball query: one block per (b,s). d2[] cached in LDS, K iterative argmins
// give the K nearest in ascending order; out-of-radius -> nearest index.
// ---------------------------------------------------------------------------
__global__ __launch_bounds__(256) void ballq_kernel(const float* __restrict__ xyz,
                                                    const float* __restrict__ nxyz,
                                                    int n, int S, int K, float radius,
                                                    int* __restrict__ gidx) {
    __shared__ float sd[8192];
    __shared__ float rv[256];
    __shared__ int   ri[256];
    __shared__ int   sel[128];
    __shared__ float seld[128];
    const int bs  = blockIdx.x;
    const int b   = bs / S;
    const int tid = threadIdx.x;
    const float* px = xyz + (size_t)b * n * 3;
    const float* q  = nxyz + (size_t)bs * 3;
    const float qx = q[0], qy = q[1], qz = q[2];
    for (int i = tid; i < n; i += 256) {
        float dx = px[i * 3 + 0] - qx;
        float dy = px[i * 3 + 1] - qy;
        float dz = px[i * 3 + 2] - qz;
        sd[i] = dx * dx + dy * dy + dz * dz;
    }
    __syncthreads();
    for (int kk = 0; kk < K; ++kk) {
        float bv = FINF; int bi = n;
        for (int i = tid; i < n; i += 256) {
            float v = sd[i];
            if (v < bv || (v == bv && i < bi)) { bv = v; bi = i; }
        }
        rv[tid] = bv; ri[tid] = bi;
        __syncthreads();
        for (int s2 = 128; s2 > 0; s2 >>= 1) {
            if (tid < s2) {
                float ov = rv[tid + s2]; int oi = ri[tid + s2];
                if (ov < rv[tid] || (ov == rv[tid] && oi < ri[tid])) { rv[tid] = ov; ri[tid] = oi; }
            }
            __syncthreads();
        }
        if (tid == 0) { sel[kk] = ri[0]; seld[kk] = rv[0]; sd[ri[0]] = FINF; }
        __syncthreads();
    }
    for (int kk = tid; kk < K; kk += 256) {
        int j = sel[kk];
        if (sqrtf(seld[kk]) > radius) j = sel[0];
        gidx[(size_t)bs * K + kk] = j;
    }
}

// ---------------------------------------------------------------------------
// Grouping: gather xyz (centered) + features into bf16 rows, channel-padded.
// ---------------------------------------------------------------------------
__global__ __launch_bounds__(256) void group_kernel(const float* __restrict__ xyz,
                                                    const float* __restrict__ nxyz,
                                                    const float* __restrict__ pts,
                                                    const int* __restrict__ gidx,
                                                    int n, int S, int K, int Cin, int Cpad,
                                                    unsigned short* __restrict__ X) {
    const int bs = blockIdx.x;
    const int b  = bs / S;
    const float* px = xyz + (size_t)b * n * 3;
    const float* q  = nxyz + (size_t)bs * 3;
    const float qx = q[0], qy = q[1], qz = q[2];
    const int* gi = gidx + (size_t)bs * K;
    const int tot = K * Cpad;
    for (int t = threadIdx.x; t < tot; t += 256) {
        const int k = t / Cpad, c = t % Cpad;
        const int j = gi[k];
        float v = 0.f;
        if (c < 3) {
            v = px[j * 3 + c] - ((c == 0) ? qx : (c == 1) ? qy : qz);
        } else if (c < 3 + Cin) {
            v = pts[((size_t)b * n + j) * Cin + (c - 3)];
        }
        X[((size_t)bs * K + k) * Cpad + c] = f2bf(v);
    }
}

// ---------------------------------------------------------------------------
// Fragment loaders per CDNA5 ISA 7.12.2 (wave32, 16-bit operands).
// A 16x32: lane half lh holds K chunks {lh*8..+7} and {lh*8+16..+7}.
// B 32x16 (from W^T): lane half lh holds K chunks {lh*8..+7} and {lh*8+16..+7}.
// ---------------------------------------------------------------------------
__device__ __forceinline__ v16bf load_a_frag(const __bf16* __restrict__ row, int k, int lh) {
    v8bf lo = *(const v8bf*)(row + k + lh * 8);
    v8bf hi = *(const v8bf*)(row + k + lh * 8 + 16);
    return __builtin_shufflevector(lo, hi,
        0, 1, 2, 3, 4, 5, 6, 7, 8, 9, 10, 11, 12, 13, 14, 15);
}
__device__ __forceinline__ v16bf load_b_frag(const __bf16* __restrict__ row, int k, int lh) {
    v8bf lo = *(const v8bf*)(row + k + lh * 8);
    v8bf hi = *(const v8bf*)(row + k + lh * 8 + 16);
    return __builtin_shufflevector(lo, hi,
        0, 1, 2, 3, 4, 5, 6, 7, 8, 9, 10, 11, 12, 13, 14, 15);
}

// ---------------------------------------------------------------------------
// GEMM: Y(MxN) = X(MxKp) * W(NxKp)^T + bias. bf16 in, f32 accumulate.
// 8 waves/block; each wave owns a 32x32 output tile = 4 accumulators,
// 4 v_wmma_f32_16x16x32_bf16 per K-step with 2x A / 2x B fragment reuse.
// Block tile: 256 rows x 32 cols. All M % 256 == 0, all Nout % 32 == 0.
// ---------------------------------------------------------------------------
__global__ __launch_bounds__(256) void gemm_bf16_wmma(const unsigned short* __restrict__ Xu,
                                                      const unsigned short* __restrict__ Wu,
                                                      const float* __restrict__ bias,
                                                      float* __restrict__ Y,
                                                      int M, int Kp, int Nout) {
    const __bf16* __restrict__ X = reinterpret_cast<const __bf16*>(Xu);
    const __bf16* __restrict__ W = reinterpret_cast<const __bf16*>(Wu);
    const int wave = threadIdx.x >> 5;
    const int lane = threadIdx.x & 31;
    const int m0 = blockIdx.x * 256 + wave * 32;
    const int n0 = blockIdx.y * 32;
    const int lr = lane & 15;    // A row / B,C column within 16-wide tile
    const int lh = lane >> 4;    // lane-half select
    const __bf16* arow0 = X + (size_t)(m0 + lr) * Kp;
    const __bf16* arow1 = X + (size_t)(m0 + 16 + lr) * Kp;
    const __bf16* wrow0 = W + (size_t)(n0 + lr) * Kp;
    const __bf16* wrow1 = W + (size_t)(n0 + 16 + lr) * Kp;
    v8f acc00 = {}, acc01 = {}, acc10 = {}, acc11 = {};
    for (int k = 0; k < Kp; k += 32) {
        __builtin_prefetch(arow0 + k + 64, 0, 3);
        __builtin_prefetch(arow1 + k + 64, 0, 3);
        __builtin_prefetch(wrow0 + k + 64, 0, 3);
        __builtin_prefetch(wrow1 + k + 64, 0, 3);
        v16bf a0 = load_a_frag(arow0, k, lh);
        v16bf a1 = load_a_frag(arow1, k, lh);
        v16bf b0 = load_b_frag(wrow0, k, lh);
        v16bf b1 = load_b_frag(wrow1, k, lh);
        acc00 = __builtin_amdgcn_wmma_f32_16x16x32_bf16(false, a0, false, b0, (short)0, acc00, false, false);
        acc01 = __builtin_amdgcn_wmma_f32_16x16x32_bf16(false, a0, false, b1, (short)0, acc01, false, false);
        acc10 = __builtin_amdgcn_wmma_f32_16x16x32_bf16(false, a1, false, b0, (short)0, acc10, false, false);
        acc11 = __builtin_amdgcn_wmma_f32_16x16x32_bf16(false, a1, false, b1, (short)0, acc11, false, false);
    }
    const int  c0  = n0 + lr;
    const int  c1  = n0 + 16 + lr;
    const float bv0 = bias[c0];
    const float bv1 = bias[c1];
#pragma unroll
    for (int i = 0; i < 8; ++i) {
        const int r0 = m0 + i + 8 * lh;       // C layout: VGPR i -> M = i (+8 hi half)
        const int r1 = r0 + 16;
        Y[(size_t)r0 * Nout + c0] = acc00[i] + bv0;
        Y[(size_t)r0 * Nout + c1] = acc01[i] + bv1;
        Y[(size_t)r1 * Nout + c0] = acc10[i] + bv0;
        Y[(size_t)r1 * Nout + c1] = acc11[i] + bv1;
    }
}

// ---------------------------------------------------------------------------
// BatchNorm statistics (population mean/var per channel over M rows).
// ---------------------------------------------------------------------------
__global__ __launch_bounds__(256) void bn_stats_kernel(const float* __restrict__ Y,
                                                       int M, int C,
                                                       float* __restrict__ stats) {
    __shared__ float s1[256], s2[256];
    const int c = blockIdx.x;
    float a = 0.f, b = 0.f;
    for (int r = threadIdx.x; r < M; r += 256) {
        float v = Y[(size_t)r * C + c];
        a += v; b += v * v;
    }
    s1[threadIdx.x] = a; s2[threadIdx.x] = b;
    __syncthreads();
    for (int s = 128; s > 0; s >>= 1) {
        if (threadIdx.x < s) { s1[threadIdx.x] += s1[threadIdx.x + s]; s2[threadIdx.x] += s2[threadIdx.x + s]; }
        __syncthreads();
    }
    if (threadIdx.x == 0) {
        float mu = s1[0] / (float)M;
        stats[c]     = mu;
        stats[C + c] = s2[0] / (float)M - mu * mu;
    }
}

__global__ void bn_apply_bf16_kernel(const float* __restrict__ Y,
                                     const float* __restrict__ stats,
                                     const float* __restrict__ g,
                                     const float* __restrict__ beta,
                                     int C, int Cpad,
                                     unsigned short* __restrict__ Xo,
                                     long long total) {
    for (long long idx = (long long)blockIdx.x * blockDim.x + threadIdx.x;
         idx < total; idx += (long long)gridDim.x * blockDim.x) {
        const int c = (int)(idx % Cpad);
        const long long r = idx / Cpad;
        float v = 0.f;
        if (c < C) {
            const float mu = stats[c];
            const float sc = g[c] * rsqrtf(stats[C + c] + 1e-5f);
            v = fmaxf(sc * (Y[r * C + c] - mu) + beta[c], 0.f);
        }
        Xo[idx] = f2bf(v);
    }
}

__global__ void bn_apply_f32_kernel(const float* __restrict__ Y,
                                    const float* __restrict__ stats,
                                    const float* __restrict__ g,
                                    const float* __restrict__ beta,
                                    int C, float* __restrict__ dst,
                                    long long total) {
    for (long long idx = (long long)blockIdx.x * blockDim.x + threadIdx.x;
         idx < total; idx += (long long)gridDim.x * blockDim.x) {
        const int c = (int)(idx % C);
        const float mu = stats[c];
        const float sc = g[c] * rsqrtf(stats[C + c] + 1e-5f);
        dst[idx] = fmaxf(sc * (Y[idx] - mu) + beta[c], 0.f);
    }
}

// ---------------------------------------------------------------------------
// Max over the K (neighbor) axis: Y is (B*S, K, C) -> P (B*S, C)
// ---------------------------------------------------------------------------
__global__ __launch_bounds__(256) void maxpool_kernel(const float* __restrict__ Y,
                                                      int K, int C,
                                                      float* __restrict__ P) {
    const int bs = blockIdx.x;
    const float* base = Y + (size_t)bs * K * C;
    for (int c = threadIdx.x; c < C; c += 256) {
        float m = -FINF;
        for (int k = 0; k < K; ++k) m = fmaxf(m, base[(size_t)k * C + c]);
        P[(size_t)bs * C + c] = m;
    }
}

// ---------------------------------------------------------------------------
// 3-NN with inverse-squared-distance weights (LDS-tiled reference set).
// ---------------------------------------------------------------------------
__global__ __launch_bounds__(256) void knn3_kernel(const float* __restrict__ x1,
                                                   const float* __restrict__ x2,
                                                   int n1, int n2,
                                                   int* __restrict__ idx3,
                                                   float* __restrict__ w3) {
    __shared__ float tx[256 * 3];
    const int b = blockIdx.y;
    const int i = blockIdx.x * 256 + threadIdx.x;
    float qx = 0.f, qy = 0.f, qz = 0.f;
    if (i < n1) {
        const float* q = x1 + ((size_t)b * n1 + i) * 3;
        qx = q[0]; qy = q[1]; qz = q[2];
    }
    float d0 = FINF, d1 = FINF, d2 = FINF;
    int   i0 = 0,    i1 = 0,    i2 = 0;
    for (int t0 = 0; t0 < n2; t0 += 256) {
        const int cnt = min(256, n2 - t0);
        __syncthreads();
        for (int t = threadIdx.x; t < cnt * 3; t += 256)
            tx[t] = x2[((size_t)b * n2 + t0) * 3 + t];
        __syncthreads();
        if (i < n1) {
            for (int j = 0; j < cnt; ++j) {
                const float dx = tx[j * 3 + 0] - qx;
                const float dy = tx[j * 3 + 1] - qy;
                const float dz = tx[j * 3 + 2] - qz;
                const float d  = dx * dx + dy * dy + dz * dz;
                const int jj = t0 + j;
                if (d < d0)      { d2 = d1; i2 = i1; d1 = d0; i1 = i0; d0 = d; i0 = jj; }
                else if (d < d1) { d2 = d1; i2 = i1; d1 = d;  i1 = jj; }
                else if (d < d2) { d2 = d;  i2 = jj; }
            }
        }
    }
    if (i < n1) {
        float w0 = 1.f / (d0 + 1e-8f);
        float w1 = 1.f / (d1 + 1e-8f);
        float w2 = 1.f / (d2 + 1e-8f);
        const float ws = w0 + w1 + w2;
        const size_t o = ((size_t)b * n1 + i) * 3;
        idx3[o] = i0; idx3[o + 1] = i1; idx3[o + 2] = i2;
        w3[o] = w0 / ws; w3[o + 1] = w1 / ws; w3[o + 2] = w2 / ws;
    }
}

// ---------------------------------------------------------------------------
// FP input build: X = concat(feat1, interp(feat2)) -> bf16 padded rows.
// ---------------------------------------------------------------------------
__global__ __launch_bounds__(256) void fpbuild_kernel(const float* __restrict__ f1, int c1,
                                                      const float* __restrict__ f2, int c2,
                                                      int n1, int n2,
                                                      const int* __restrict__ idx3,
                                                      const float* __restrict__ w3,
                                                      int Cpad,
                                                      unsigned short* __restrict__ X) {
    const int row = blockIdx.x;            // b*n1 + i
    const size_t o = (size_t)row * 3;
    const int j0 = idx3[o], j1 = idx3[o + 1], j2 = idx3[o + 2];
    const float w0 = w3[o], w1 = w3[o + 1], w2 = w3[o + 2];
    const int b = row / n1;
    const float* F2 = f2 + (size_t)b * n2 * c2;
    for (int c = threadIdx.x; c < Cpad; c += 256) {
        float v = 0.f;
        if (c < c1) {
            v = f1[(size_t)row * c1 + c];
        } else if (c < c1 + c2) {
            const int cc = c - c1;
            v = w0 * F2[(size_t)j0 * c2 + cc]
              + w1 * F2[(size_t)j1 * c2 + cc]
              + w2 * F2[(size_t)j2 * c2 + cc];
        }
        X[(size_t)row * Cpad + c] = f2bf(v);
    }
}

// ---------------------------------------------------------------------------
// Weight prep: fp32 (Cout,Cin) -> bf16 (Cout,Cpad), zero-padded K.
// ---------------------------------------------------------------------------
__global__ void wprep_kernel(const float* __restrict__ W, int Cin, int Cpad,
                             long long total, unsigned short* __restrict__ Wp) {
    for (long long idx = (long long)blockIdx.x * blockDim.x + threadIdx.x;
         idx < total; idx += (long long)gridDim.x * blockDim.x) {
        const int c = (int)(idx % Cpad);
        const long long o = idx / Cpad;
        Wp[idx] = f2bf(c < Cin ? W[o * Cin + c] : 0.f);
    }
}

// Final (B*N, C) -> (B, C, N)
__global__ void transpose_out_kernel(const float* __restrict__ X, int N, int C,
                                     float* __restrict__ out, long long total) {
    for (long long idx = (long long)blockIdx.x * blockDim.x + threadIdx.x;
         idx < total; idx += (long long)gridDim.x * blockDim.x) {
        const int i = (int)(idx % N);
        const long long t = idx / N;
        const int c = (int)(t % C);
        const int b = (int)(t / C);
        out[idx] = X[((size_t)b * N + i) * C + c];
    }
}

// ---------------------------------------------------------------------------
extern "C" void kernel_launch(void* const* d_in, const int* in_sizes, int n_in,
                              void* d_out, int out_size, void* d_ws, size_t ws_size,
                              hipStream_t stream) {
    (void)in_sizes; (void)n_in; (void)out_size; (void)ws_size;
    constexpr int B = 2, N = 8192;
    constexpr int NL = 15;
    static const int CIN[NL]  = {3, 64, 64,   131, 128, 128,   259, 256, 256,   768, 512,   384, 256,   128, 128};
    static const int COUT[NL] = {64, 64, 128, 128, 128, 256,   256, 256, 512,   512, 256,   256, 128,   128, 64};
    int CPAD[NL];
    for (int l = 0; l < NL; ++l) CPAD[l] = (CIN[l] + 31) & ~31;

    const float* xyz = (const float*)d_in[0];
    const float *Wf[NL], *Bf[NL], *Gf[NL], *Tf[NL];
    {
        int p = 1;
        for (int l = 0; l < NL; ++l) {
            Wf[l] = (const float*)d_in[p++];
            Bf[l] = (const float*)d_in[p++];
            Gf[l] = (const float*)d_in[p++];
            Tf[l] = (const float*)d_in[p++];
        }
    }

    // ---- workspace bump allocator -----------------------------------------
    char* ws = (char*)d_ws;
    size_t off = 0;
    auto alloc = [&](size_t bytes) -> void* {
        void* p = ws + off;
        off = (off + bytes + 255) & ~(size_t)255;
        return p;
    };
    unsigned short* Wp[NL];
    for (int l = 0; l < NL; ++l) Wp[l] = (unsigned short*)alloc((size_t)COUT[l] * CPAD[l] * 2);

    float* l1x = (float*)alloc((size_t)B * 4096 * 3 * 4);
    float* l2x = (float*)alloc((size_t)B * 1024 * 3 * 4);
    float* l3x = (float*)alloc((size_t)B * 512  * 3 * 4);
    int* gidx1 = (int*)alloc((size_t)B * 4096 * 32  * 4);
    int* gidx2 = (int*)alloc((size_t)B * 1024 * 64  * 4);
    int* gidx3 = (int*)alloc((size_t)B * 512  * 128 * 4);
    float* l1p = (float*)alloc((size_t)B * 4096 * 128 * 4);
    float* l2p = (float*)alloc((size_t)B * 1024 * 256 * 4);
    float* l3p = (float*)alloc((size_t)B * 512  * 512 * 4);
    float* f2u = (float*)alloc((size_t)B * 1024 * 256 * 4);
    float* f1u = (float*)alloc((size_t)B * 4096 * 128 * 4);
    int*   idx3 = (int*)alloc((size_t)B * 8192 * 3 * 4);
    float* w3   = (float*)alloc((size_t)B * 8192 * 3 * 4);
    float* stats = (float*)alloc(1024 * 4);
    const size_t XMAX = (size_t)131072 * 288;           // largest bf16 activation
    unsigned short* XA = (unsigned short*)alloc(XMAX * 2);
    unsigned short* XB = (unsigned short*)alloc(XMAX * 2);
    float* Y = (float*)alloc((size_t)131072 * 512 * 4); // largest fp32 GEMM out

    // ---- weight prep ------------------------------------------------------
    for (int l = 0; l < NL; ++l) {
        long long tot = (long long)COUT[l] * CPAD[l];
        wprep_kernel<<<EW_BLOCKS, 256, 0, stream>>>(Wf[l], CIN[l], CPAD[l], tot, Wp[l]);
    }

    auto run_gemm = [&](const unsigned short* X, int l, int M) {
        dim3 g(M / 256, COUT[l] / 32);
        gemm_bf16_wmma<<<g, 256, 0, stream>>>(X, Wp[l], Bf[l], Y, M, CPAD[l], COUT[l]);
        bn_stats_kernel<<<COUT[l], 256, 0, stream>>>(Y, M, COUT[l], stats);
    };
    auto apply_bf16 = [&](int l, int M, int CpadN, unsigned short* Xn) {
        long long tot = (long long)M * CpadN;
        bn_apply_bf16_kernel<<<EW_BLOCKS, 256, 0, stream>>>(Y, stats, Gf[l], Tf[l],
                                                            COUT[l], CpadN, Xn, tot);
    };
    auto apply_f32 = [&](int l, int M, float* dst) {
        long long tot = (long long)M * COUT[l];
        bn_apply_f32_kernel<<<EW_BLOCKS, 256, 0, stream>>>(Y, stats, Gf[l], Tf[l],
                                                           COUT[l], dst, tot);
    };

    // ---- SA1: n=8192 -> S=4096, K=32, r=0.2, layers 0..2 ------------------
    fps_kernel<<<B, 1024, 0, stream>>>(xyz, 8192, 4096, l1x);
    ballq_kernel<<<B * 4096, 256, 0, stream>>>(xyz, l1x, 8192, 4096, 32, 0.2f, gidx1);
    group_kernel<<<B * 4096, 256, 0, stream>>>(xyz, l1x, nullptr, gidx1, 8192, 4096, 32, 0, CPAD[0], XA);
    {
        const int M = B * 4096 * 32;
        run_gemm(XA, 0, M); apply_bf16(0, M, CPAD[1], XB);
        run_gemm(XB, 1, M); apply_bf16(1, M, CPAD[2], XA);
        run_gemm(XA, 2, M); apply_f32(2, M, Y);
        maxpool_kernel<<<B * 4096, 256, 0, stream>>>(Y, 32, 128, l1p);
    }

    // ---- SA2: n=4096 -> S=1024, K=64, r=0.4, layers 3..5 ------------------
    fps_kernel<<<B, 1024, 0, stream>>>(l1x, 4096, 1024, l2x);
    ballq_kernel<<<B * 1024, 256, 0, stream>>>(l1x, l2x, 4096, 1024, 64, 0.4f, gidx2);
    group_kernel<<<B * 1024, 256, 0, stream>>>(l1x, l2x, l1p, gidx2, 4096, 1024, 64, 128, CPAD[3], XA);
    {
        const int M = B * 1024 * 64;
        run_gemm(XA, 3, M); apply_bf16(3, M, CPAD[4], XB);
        run_gemm(XB, 4, M); apply_bf16(4, M, CPAD[5], XA);
        run_gemm(XA, 5, M); apply_f32(5, M, Y);
        maxpool_kernel<<<B * 1024, 256, 0, stream>>>(Y, 64, 256, l2p);
    }

    // ---- SA3: n=1024 -> S=512, K=128, r=0.6, layers 6..8 ------------------
    fps_kernel<<<B, 1024, 0, stream>>>(l2x, 1024, 512, l3x);
    ballq_kernel<<<B * 512, 256, 0, stream>>>(l2x, l3x, 1024, 512, 128, 0.6f, gidx3);
    group_kernel<<<B * 512, 256, 0, stream>>>(l2x, l3x, l2p, gidx3, 1024, 512, 128, 256, CPAD[6], XA);
    {
        const int M = B * 512 * 128;
        run_gemm(XA, 6, M); apply_bf16(6, M, CPAD[7], XB);
        run_gemm(XB, 7, M); apply_bf16(7, M, CPAD[8], XA);
        run_gemm(XA, 8, M); apply_f32(8, M, Y);
        maxpool_kernel<<<B * 512, 256, 0, stream>>>(Y, 128, 512, l3p);
    }

    // ---- FP3: l2x(1024) <- l3x(512); concat(l2p[256], interp l3p[512]) ----
    knn3_kernel<<<dim3(1024 / 256, B), 256, 0, stream>>>(l2x, l3x, 1024, 512, idx3, w3);
    fpbuild_kernel<<<B * 1024, 256, 0, stream>>>(l2p, 256, l3p, 512, 1024, 512, idx3, w3, CPAD[9], XA);
    {
        const int M = B * 1024;
        run_gemm(XA, 9, M);  apply_bf16(9, M, CPAD[10], XB);
        run_gemm(XB, 10, M); apply_f32(10, M, f2u);
    }

    // ---- FP2: l1x(4096) <- l2x(1024); concat(l1p[128], interp f2u[256]) ---
    knn3_kernel<<<dim3(4096 / 256, B), 256, 0, stream>>>(l1x, l2x, 4096, 1024, idx3, w3);
    fpbuild_kernel<<<B * 4096, 256, 0, stream>>>(l1p, 128, f2u, 256, 4096, 1024, idx3, w3, CPAD[11], XA);
    {
        const int M = B * 4096;
        run_gemm(XA, 11, M); apply_bf16(11, M, CPAD[12], XB);
        run_gemm(XB, 12, M); apply_f32(12, M, f1u);
    }

    // ---- FP1: xyz(8192) <- l1x(4096); interp f1u[128] only ----------------
    knn3_kernel<<<dim3(8192 / 256, B), 256, 0, stream>>>(xyz, l1x, 8192, 4096, idx3, w3);
    fpbuild_kernel<<<B * 8192, 256, 0, stream>>>(nullptr, 0, f1u, 128, 8192, 4096, idx3, w3, CPAD[13], XA);
    {
        const int M = B * 8192;
        run_gemm(XA, 13, M); apply_bf16(13, M, CPAD[14], XB);
        run_gemm(XB, 14, M); apply_f32(14, M, Y);   // (B*N, 64), BN+ReLU applied
    }
    transpose_out_kernel<<<EW_BLOCKS, 256, 0, stream>>>(Y, N, 64, (float*)d_out,
                                                        (long long)B * 64 * N);
}
